// VisionTransformerBlock_16226386444700
// MI455X (gfx1250) — matrix operand order repaired and checked by reading, not compiled, not run on hardware
//
#include <hip/hip_runtime.h>

typedef __bf16 bf16_t;
typedef __attribute__((ext_vector_type(16))) __bf16 v16bf;
typedef __attribute__((ext_vector_type(8)))  __bf16 v8bf;
typedef __attribute__((ext_vector_type(8)))  float  v8f;

#define HEADS  8
#define PATCH  256
#define BMROWS 192                  // B*M = 32*6
#define ROWS   (BMROWS*PATCH)       // 49152 token rows
#define MROWS  (BMROWS*128)         // 24576 merged rows
#define LN_EPS 1e-5f

// ---- cross-lane xor shuffle within 32-lane group via ds_swizzle ----
#define SWZ_XOR(v, m) __int_as_float(__builtin_amdgcn_ds_swizzle(__float_as_int(v), (((m) << 10) | 0x1F)))

// A-matrix 16x32 bf16 fragment: lane<16 holds K={0..7,16..23}, lane>=16 K={8..15,24..31}
static __device__ __forceinline__ v16bf load_a_frag(const bf16_t* row, int k0, int half) {
  const v8bf c0 = *(const v8bf*)(row + k0 + 8 * half);
  const v8bf c1 = *(const v8bf*)(row + k0 + 8 * half + 16);
  v16bf r;
#pragma unroll
  for (int i = 0; i < 8; ++i) { r[i] = c0[i]; r[i + 8] = c1[i]; }
  return r;
}

// B-matrix 32x16 bf16 fragment: lane<16 holds K={0..7,16..23}, lane>=16 K={8..15,24..31}
static __device__ __forceinline__ v16bf load_b_frag(const bf16_t* row, int k0, int half) {
  const v8bf c0 = *(const v8bf*)(row + k0 + 8 * half);
  const v8bf c1 = *(const v8bf*)(row + k0 + 8 * half + 16);
  v16bf r;
#pragma unroll
  for (int i = 0; i < 8; ++i) { r[i] = c0[i]; r[i + 8] = c1[i]; }
  return r;
}

static __device__ __forceinline__ v8f wmma_bf16(v16bf a, v16bf b, v8f c) {
  return __builtin_amdgcn_wmma_f32_16x16x32_bf16(false, a, false, b, (short)0, c, false, false);
}

// -------------------- conversion kernels --------------------
__global__ __launch_bounds__(256) void convert_bf16_kernel(const float* __restrict__ in,
                                                           bf16_t* __restrict__ out, int n) {
  int i = (blockIdx.x * 256 + threadIdx.x) * 4;
  if (i >= n) return;
  float4 v = *(const float4*)(in + i);
  out[i + 0] = (bf16_t)v.x;
  out[i + 1] = (bf16_t)v.y;
  out[i + 2] = (bf16_t)v.z;
  out[i + 3] = (bf16_t)v.w;
}

// out[n*K + k] = (bf16) in[k*N + n]   (weight [K,N] -> Bt [N,K])
__global__ __launch_bounds__(256) void transpose_bf16_kernel(const float* __restrict__ in,
                                                             bf16_t* __restrict__ out,
                                                             int K, int N) {
  int idx = blockIdx.x * 256 + threadIdx.x;
  if (idx >= K * N) return;
  int k = idx / N, n = idx % N;
  out[(size_t)n * K + k] = (bf16_t)in[idx];
}

// -------------------- generic bf16 WMMA GEMM --------------------
// C[M,N] = A[M,KDIM] @ Bt[N,KDIM]^T + bias ; one wave computes a 32x64 tile
// (2 M-tiles x 4 N-tiles, B fragments reused across both M-tiles; K fully unrolled).
// MODE 0: f32 row-major.  MODE 1: bf16 -> [bm,h,p,a].  MODE 2: bf16 -> [bm,h,a,p].
template <int MODE, int KDIM>
__global__ __launch_bounds__(256) void gemm_bf16_kernel(const bf16_t* __restrict__ Abf,
                                                        const bf16_t* __restrict__ Bt,
                                                        const float* __restrict__ bias,
                                                        void* __restrict__ outv,
                                                        int Mrows, int N) {
  const int wave = (blockIdx.x * 256 + threadIdx.x) >> 5;
  const int lane = threadIdx.x & 31;
  const int half = lane >> 4;
  const int l = lane & 15;
  const int ntile4 = N >> 6;
  const int tm = wave / ntile4;
  const int tn = wave % ntile4;
  const int m0 = tm * 32;
  if (m0 >= Mrows) return;
  const int n0 = tn * 64;

  const bf16_t* arow0 = Abf + (size_t)(m0 + l) * KDIM;
  const bf16_t* arow1 = arow0 + (size_t)16 * KDIM;
  const bf16_t* brow[4];
#pragma unroll
  for (int j = 0; j < 4; ++j) brow[j] = Bt + (size_t)(n0 + 16 * j + l) * KDIM;

  v8f acc0[4] = {};
  v8f acc1[4] = {};
#pragma unroll
  for (int k0 = 0; k0 < KDIM; k0 += 32) {
    const v16bf a0 = load_a_frag(arow0, k0, half);
    const v16bf a1 = load_a_frag(arow1, k0, half);
#pragma unroll
    for (int j = 0; j < 4; ++j) {
      const v16bf b = load_b_frag(brow[j], k0, half);
      acc0[j] = wmma_bf16(a0, b, acc0[j]);
      acc1[j] = wmma_bf16(a1, b, acc1[j]);
    }
  }

#pragma unroll
  for (int j = 0; j < 4; ++j) {
    const int col = n0 + 16 * j + l;
    const float bc = bias[col];
#pragma unroll
    for (int mt = 0; mt < 2; ++mt) {
      const v8f acc = mt ? acc1[j] : acc0[j];
#pragma unroll
      for (int r = 0; r < 8; ++r) {
        const int row = m0 + mt * 16 + r + 8 * half;
        const float val = acc[r] + bc;
        if (MODE == 0) {
          ((float*)outv)[(size_t)row * N + col] = val;
        } else if (MODE == 1) {
          const int bm = row >> 8, p = row & 255;
          const int h = col >> 6, a = col & 63;
          ((bf16_t*)outv)[(size_t)(bm * HEADS + h) * 16384 + p * 64 + a] = (bf16_t)val;
        } else {
          const int bm = row >> 8, p = row & 255;
          const int h = col >> 6, a = col & 63;
          ((bf16_t*)outv)[(size_t)(bm * HEADS + h) * 16384 + a * 256 + p] = (bf16_t)val;
        }
      }
    }
  }
}

// -------------------- fused attention --------------------
// Q,K: [bm,h,p,a] bf16 ; Vt: [bm,h,a,p] bf16 ; Z: [row, h*64+a] bf16
// block = 4 waves; blockIdx.x = bmh*4 + qblock ; wave handles 16 query rows.
__global__ __launch_bounds__(128) void attn_kernel(const bf16_t* __restrict__ Q,
                                                   const bf16_t* __restrict__ Kb,
                                                   const bf16_t* __restrict__ Vt,
                                                   bf16_t* __restrict__ Z) {
  __shared__ alignas(32) bf16_t lds_attn[4 * 16 * 256];  // 32 KB

  const int bmh = blockIdx.x >> 2;
  const int qb = blockIdx.x & 3;
  const int w = threadIdx.x >> 5;
  const int lane = threadIdx.x & 31;
  const int half = lane >> 4;
  const int l = lane & 15;
  const int qr0 = qb * 64 + w * 16;

  const bf16_t* qbase = Q + (size_t)bmh * 16384;
  const bf16_t* kbase = Kb + (size_t)bmh * 16384;
  const bf16_t* vbase = Vt + (size_t)bmh * 16384;

  // q A-fragments for K=64 (two 32-wide steps)
  const bf16_t* qrow = qbase + (size_t)(qr0 + l) * 64;
  const v16bf aq0 = load_a_frag(qrow, 0, half);
  const v16bf aq1 = load_a_frag(qrow, 32, half);

  // scores: 16 key tiles of 16 columns each
  v8f sc[16];
#pragma unroll
  for (int t = 0; t < 16; ++t) {
    const bf16_t* krow = kbase + (size_t)(t * 16 + l) * 64;
    v8f c = {};
    c = wmma_bf16(aq0, load_b_frag(krow, 0, half), c);
    c = wmma_bf16(aq1, load_b_frag(krow, 32, half), c);
    sc[t] = c;
  }

  // softmax over 256 columns per row (16 tiles x 16 lanes in half-wave)
  const float scale = 0.125f;  // 1/sqrt(64)
#pragma unroll
  for (int r = 0; r < 8; ++r) {
    float mx = -1e30f;
#pragma unroll
    for (int t = 0; t < 16; ++t) { sc[t][r] *= scale; mx = fmaxf(mx, sc[t][r]); }
    mx = fmaxf(mx, SWZ_XOR(mx, 1));
    mx = fmaxf(mx, SWZ_XOR(mx, 2));
    mx = fmaxf(mx, SWZ_XOR(mx, 4));
    mx = fmaxf(mx, SWZ_XOR(mx, 8));
    float sum = 0.f;
#pragma unroll
    for (int t = 0; t < 16; ++t) { float e = __expf(sc[t][r] - mx); sc[t][r] = e; sum += e; }
    sum += SWZ_XOR(sum, 1);
    sum += SWZ_XOR(sum, 2);
    sum += SWZ_XOR(sum, 4);
    sum += SWZ_XOR(sum, 8);
    const float inv = 1.0f / sum;
#pragma unroll
    for (int t = 0; t < 16; ++t) sc[t][r] *= inv;
  }

  // C-layout -> A-layout reshuffle through LDS
  bf16_t* la = lds_attn + w * 16 * 256;
#pragma unroll
  for (int t = 0; t < 16; ++t)
#pragma unroll
    for (int r = 0; r < 8; ++r)
      la[(r + 8 * half) * 256 + t * 16 + l] = (bf16_t)sc[t][r];
  __syncthreads();

  // z = attn @ v  (M=16, N=64, K=256)
  const int bm = bmh >> 3, h = bmh & 7;
  const bf16_t* arow = la + (size_t)l * 256;
#pragma unroll
  for (int j = 0; j < 4; ++j) {
    const bf16_t* vrow = vbase + (size_t)(j * 16 + l) * 256;
    v8f c = {};
#pragma unroll
    for (int ks = 0; ks < 8; ++ks)
      c = wmma_bf16(load_a_frag(arow, ks * 32, half), load_b_frag(vrow, ks * 32, half), c);
#pragma unroll
    for (int r = 0; r < 8; ++r) {
      const int p = qr0 + r + 8 * half;
      const int col = h * 64 + j * 16 + l;
      Z[((size_t)bm * 256 + p) * 512 + col] = (bf16_t)c[r];
    }
  }
}

// -------------------- residual + LN + ReLU + patch merge --------------------
__global__ __launch_bounds__(256) void ln_merge_kernel(const float* __restrict__ y,
                                                       const float* __restrict__ x,
                                                       const float* __restrict__ g1,
                                                       const float* __restrict__ b1,
                                                       bf16_t* __restrict__ merged) {
  __shared__ float red[256];
  const int row = blockIdx.x;
  const int t = threadIdx.x;
  const float* yr = y + (size_t)row * 512;
  const float v0 = yr[t], v1 = yr[t + 256];

  red[t] = v0 + v1;
  __syncthreads();
  for (int o = 128; o > 0; o >>= 1) { if (t < o) red[t] += red[t + o]; __syncthreads(); }
  const float mu = red[0] * (1.0f / 512.0f);
  __syncthreads();
  const float d0 = v0 - mu, d1 = v1 - mu;
  red[t] = d0 * d0 + d1 * d1;
  __syncthreads();
  for (int o = 128; o > 0; o >>= 1) { if (t < o) red[t] += red[t + o]; __syncthreads(); }
  const float rs = rsqrtf(red[0] * (1.0f / 512.0f) + LN_EPS);

  const int bm = row >> 8, p = row & 255;
  bf16_t* mrow = merged + ((size_t)bm * 128 + (p >> 1)) * 1024 + (p & 1) * 512;
#pragma unroll
  for (int e = 0; e < 2; ++e) {
    const int c = t + e * 256;
    const float v = (e == 0) ? v0 : v1;
    const float ln = (v - mu) * rs * g1[c] + b1[c];
    mrow[c] = (bf16_t)fmaxf(0.f, x[(size_t)row * 512 + c] + ln);
  }
}

// -------------------- final LN + ReLU --------------------
__global__ __launch_bounds__(256) void ln_final_kernel(const float* __restrict__ y,
                                                       const float* __restrict__ g2,
                                                       const float* __restrict__ b2,
                                                       float* __restrict__ out) {
  __shared__ float red[256];
  const int row = blockIdx.x;
  const int t = threadIdx.x;
  const float* yr = y + (size_t)row * 512;
  const float v0 = yr[t], v1 = yr[t + 256];

  red[t] = v0 + v1;
  __syncthreads();
  for (int o = 128; o > 0; o >>= 1) { if (t < o) red[t] += red[t + o]; __syncthreads(); }
  const float mu = red[0] * (1.0f / 512.0f);
  __syncthreads();
  const float d0 = v0 - mu, d1 = v1 - mu;
  red[t] = d0 * d0 + d1 * d1;
  __syncthreads();
  for (int o = 128; o > 0; o >>= 1) { if (t < o) red[t] += red[t + o]; __syncthreads(); }
  const float rs = rsqrtf(red[0] * (1.0f / 512.0f) + LN_EPS);

  out[(size_t)row * 512 + t] = fmaxf(0.f, (v0 - mu) * rs * g2[t] + b2[t]);
  out[(size_t)row * 512 + t + 256] = fmaxf(0.f, (v1 - mu) * rs * g2[t + 256] + b2[t + 256]);
}

// -------------------- launcher --------------------
extern "C" void kernel_launch(void* const* d_in, const int* in_sizes, int n_in,
                              void* d_out, int out_size, void* d_ws, size_t ws_size,
                              hipStream_t stream) {
  (void)in_sizes; (void)n_in; (void)out_size;
  const float* x  = (const float*)d_in[0];
  const float* Wq = (const float*)d_in[1];
  const float* bq = (const float*)d_in[2];
  const float* Wk = (const float*)d_in[3];
  const float* bk = (const float*)d_in[4];
  const float* Wv = (const float*)d_in[5];
  const float* bv = (const float*)d_in[6];
  const float* Wo = (const float*)d_in[7];
  const float* bo = (const float*)d_in[8];
  const float* g1 = (const float*)d_in[9];
  const float* b1 = (const float*)d_in[10];
  const float* Wd = (const float*)d_in[11];
  const float* bd = (const float*)d_in[12];
  const float* g2 = (const float*)d_in[13];
  const float* b2 = (const float*)d_in[14];

  const size_t NEED = 355467264ull;
  if (ws_size < NEED) return;

  char* ws = (char*)d_ws;
  bf16_t* x_bf  = (bf16_t*)(ws);                    // 49152x512 bf16 (reused as merged)
  bf16_t* Wq_t  = (bf16_t*)(ws + 50331648ull);      // [512][512]
  bf16_t* Wk_t  = Wq_t + 512 * 512;
  bf16_t* Wv_t  = Wk_t + 512 * 512;
  bf16_t* Wo_t  = Wv_t + 512 * 512;
  bf16_t* Wd_t  = Wo_t + 512 * 512;                 // [512][1024]
  bf16_t* q_bf  = (bf16_t*)(ws + 53477376ull);      // [bm,h,p,a] (reused as wd_out f32)
  bf16_t* k_bf  = (bf16_t*)(ws + 103809024ull);
  bf16_t* vT_bf = (bf16_t*)(ws + 154140672ull);     // [bm,h,a,p]
  bf16_t* z_bf  = (bf16_t*)(ws + 204472320ull);     // [49152,512]
  float*  wo_out = (float*)(ws + 254803968ull);     // [49152,512] f32
  bf16_t* merged = x_bf;                            // [24576,1024] bf16
  float*  wd_out = (float*)q_bf;                    // [24576,512] f32

  // conversions
  convert_bf16_kernel<<<(ROWS * 512 / 4 + 255) / 256, 256, 0, stream>>>(x, x_bf, ROWS * 512);
  transpose_bf16_kernel<<<(512 * 512 + 255) / 256, 256, 0, stream>>>(Wq, Wq_t, 512, 512);
  transpose_bf16_kernel<<<(512 * 512 + 255) / 256, 256, 0, stream>>>(Wk, Wk_t, 512, 512);
  transpose_bf16_kernel<<<(512 * 512 + 255) / 256, 256, 0, stream>>>(Wv, Wv_t, 512, 512);
  transpose_bf16_kernel<<<(512 * 512 + 255) / 256, 256, 0, stream>>>(Wo, Wo_t, 512, 512);
  transpose_bf16_kernel<<<(1024 * 512 + 255) / 256, 256, 0, stream>>>(Wd, Wd_t, 1024, 512);

  // QKV projections: waves = (49152/32)*(512/64) = 12288 -> 1536 blocks
  gemm_bf16_kernel<1, 512><<<1536, 256, 0, stream>>>(x_bf, Wq_t, bq, q_bf, ROWS, 512);
  gemm_bf16_kernel<1, 512><<<1536, 256, 0, stream>>>(x_bf, Wk_t, bk, k_bf, ROWS, 512);
  gemm_bf16_kernel<2, 512><<<1536, 256, 0, stream>>>(x_bf, Wv_t, bv, vT_bf, ROWS, 512);

  // attention: 1536 (b,m,h) x 4 query-blocks
  attn_kernel<<<BMROWS * HEADS * 4, 128, 0, stream>>>(q_bf, k_bf, vT_bf, z_bf);

  // output projection
  gemm_bf16_kernel<0, 512><<<1536, 256, 0, stream>>>(z_bf, Wo_t, bo, wo_out, ROWS, 512);

  // residual + LN + ReLU + patch merge
  ln_merge_kernel<<<ROWS, 256, 0, stream>>>(wo_out, x, g1, b1, merged);

  // dense merge projection: waves = (24576/32)*(512/64) = 6144 -> 768 blocks
  gemm_bf16_kernel<0, 1024><<<768, 256, 0, stream>>>(merged, Wd_t, bd, wd_out, MROWS, 512);

  // final LN + ReLU
  ln_final_kernel<<<MROWS, 256, 0, stream>>>(wd_out, g2, b2, (float*)d_out);
}
